// ATKT_15152644620347
// MI455X (gfx1250) — matrix operand order repaired and checked by reading, not compiled, not run on hardware
//
#include <hip/hip_runtime.h>
#include <hip/hip_bf16.h>
#include <math.h>

// Problem constants (match reference)
#define BB   64
#define SS   1024
#define DCc  64
#define Hh   128
#define G3   384         // 3*H
#define BS   (BB*SS)

typedef __bf16 bf16_t;
typedef __attribute__((ext_vector_type(16))) __bf16 v16bf;
typedef __attribute__((ext_vector_type(8)))  float  v8f;

__device__ __forceinline__ bf16_t f2bf(float f) {
  union { float f; unsigned u; } a; a.f = f;
  unsigned r = a.u + 0x7FFFu + ((a.u >> 16) & 1u);
  unsigned short h = (unsigned short)(r >> 16);
  bf16_t b; __builtin_memcpy(&b, &h, 2); return b;
}
__device__ __forceinline__ float sigm(float x) { return 1.0f / (1.0f + __expf(-x)); }

// A-matrix (16x32 bf16) per-lane load: lanes 0-15 hold K {0..15}, lanes 16-31 hold K {16..31}
// base = row_ptr + kk*32 + half*16
__device__ __forceinline__ v16bf load_a_bf(const bf16_t* base) {
  v16bf a;
#pragma unroll
  for (int i = 0; i < 16; ++i) a[i] = base[i];
  return a;
}
// B-matrix (32x16 bf16) per-lane load: lane holds col N=lane&15, K {0..15} (half=0) or {16..31} (half=1)
// base = col_ptr + kk*32 + half*16  (column contiguous in memory == weight row)
__device__ __forceinline__ v16bf load_b_bf(const bf16_t* base) {
  v16bf b;
#pragma unroll
  for (int i = 0; i < 16; ++i) b[i] = base[i];
  return b;
}

// ---- kernel 1: convert W_ih, W_hh, mlp_w to bf16 ----
__global__ void k_cvt(const float* __restrict__ Wih, const float* __restrict__ Whh,
                      const float* __restrict__ mlpw,
                      bf16_t* __restrict__ Wihb, bf16_t* __restrict__ Whhb,
                      bf16_t* __restrict__ mlpwb) {
  int i = blockIdx.x * blockDim.x + threadIdx.x;
  if (i < G3 * Hh) { Wihb[i] = f2bf(Wih[i]); Whhb[i] = f2bf(Whh[i]); }
  if (i < Hh * Hh) { mlpwb[i] = f2bf(mlpw[i]); }
}

// ---- kernel 2: fused concept embedding (masked mean) + inter construction ----
// one wave per (b,s); lane owns 2 of 64 concept dims
__global__ void k_embed(const int* __restrict__ qseq, const int* __restrict__ cseq,
                        const int* __restrict__ q2c, const int* __restrict__ q2cm,
                        const float* __restrict__ cemb, const float* __restrict__ remb,
                        float* __restrict__ concept, bf16_t* __restrict__ interb) {
  int widx = (blockIdx.x * blockDim.x + threadIdx.x) >> 5;
  int lane = threadIdx.x & 31;
  int q = qseq[widx];
  float msum = 0.f, c0 = 0.f, c1 = 0.f;
#pragma unroll
  for (int mc = 0; mc < 4; ++mc) {
    int cid = q2c[q * 4 + mc];
    float mk = (float)q2cm[q * 4 + mc];
    msum += mk;
    c0 += mk * cemb[cid * 64 + lane * 2 + 0];
    c1 += mk * cemb[cid * 64 + lane * 2 + 1];
  }
  float inv = 1.f / fmaxf(msum, 1.f);
  c0 *= inv; c1 *= inv;
  concept[(size_t)widx * 64 + lane * 2 + 0] = c0;
  concept[(size_t)widx * 64 + lane * 2 + 1] = c1;
  int corr = cseq[widx];
  float r0 = remb[corr * 64 + lane * 2 + 0];
  float r1 = remb[corr * 64 + lane * 2 + 1];
  float f0 = corr ? c0 : r0, f1 = corr ? c1 : r1;
  float s0 = corr ? r0 : c0, s1 = corr ? r1 : c1;
  bf16_t* ib = interb + (size_t)widx * 128;
  ib[lane * 2 + 0] = f2bf(f0); ib[lane * 2 + 1] = f2bf(f1);
  ib[64 + lane * 2 + 0] = f2bf(s0); ib[64 + lane * 2 + 1] = f2bf(s1);
}

// ---- kernel 3: Xp = inter @ W_ih^T + b_ih  via bf16 WMMA (65536 x 384 x 128) ----
// one wave per 16-row M-tile; A tile stays in VGPRs across all 24 N-tiles
__global__ void k_xp(const bf16_t* __restrict__ interb, const bf16_t* __restrict__ Wihb,
                     const float* __restrict__ bih, float* __restrict__ Xp) {
  int mt = (blockIdx.x * blockDim.x + threadIdx.x) >> 5;
  int lane = threadIdx.x & 31;
  int m = lane & 15, half = lane >> 4;

  v16bf A[4];
  const bf16_t* arow = interb + (size_t)(mt * 16 + m) * 128 + half * 16;
#pragma unroll
  for (int kk = 0; kk < 4; ++kk) A[kk] = load_a_bf(arow + kk * 32);

  int rbase = mt * 16 + half * 8;
  for (int nt = 0; nt < 24; ++nt) {
    int n = nt * 16 + m;
    const bf16_t* bcol = Wihb + (size_t)n * 128 + half * 16;
    v8f acc = {0.f, 0.f, 0.f, 0.f, 0.f, 0.f, 0.f, 0.f};
#pragma unroll
    for (int kk = 0; kk < 4; ++kk) {
      v16bf Bm = load_b_bf(bcol + kk * 32);
      acc = __builtin_amdgcn_wmma_f32_16x16x32_bf16(false, A[kk], false, Bm, (short)0, acc, false, false);
    }
    float bias = bih[n];
#pragma unroll
    for (int r = 0; r < 8; ++r) Xp[(size_t)(rbase + r) * G3 + n] = acc[r] + bias;
  }
}

// ---- kernel 4: GRU scan. 4 WGPs x 16 batch rows; W_hh lives in VGPRs, h in LDS ----
// Per step: issue this step's Xp global loads FIRST (latency hidden behind WMMA phase
// + barrier), prefetch next step's lines into cache, then WMMA, then gate math.
__global__ void __launch_bounds__(256) k_gru(const float* __restrict__ Xp,
                                             const bf16_t* __restrict__ Whhb,
                                             const float* __restrict__ bhh,
                                             float* __restrict__ rnnf,
                                             bf16_t* __restrict__ rnnb) {
  __shared__ bf16_t hl[16 * 128];  // h in bf16 (WMMA A operand)
  __shared__ float  hf[16 * 128];  // h in f32 (gate math)
  __shared__ float  ghl[16 * G3];  // gh tile
  const int tid = threadIdx.x;
  const int wave = tid >> 5, lane = tid & 31;
  const int m = lane & 15, half = lane >> 4;
  const int bb = blockIdx.x * 16;
  const int grow = tid >> 4;          // gate-stage row 0..15
  const int gd0  = (tid & 15) * 8;    // gate-stage dim base

  // preload this wave's 3 N-tiles of W_hh^T into registers (12 WMMA B operands)
  v16bf Breg[3][4];
#pragma unroll
  for (int j = 0; j < 3; ++j) {
    int n = (wave * 3 + j) * 16 + m;
    const bf16_t* bcol = Whhb + (size_t)n * 128 + half * 16;
#pragma unroll
    for (int kk = 0; kk < 4; ++kk) Breg[j][kk] = load_b_bf(bcol + kk * 32);
  }
  // per-thread gate biases (loop-invariant)
  float bhr[8], bhz[8], bhn[8];
#pragma unroll
  for (int i = 0; i < 8; ++i) {
    bhr[i] = bhh[gd0 + i];
    bhz[i] = bhh[128 + gd0 + i];
    bhn[i] = bhh[256 + gd0 + i];
  }
  for (int e = tid; e < 16 * 128; e += 256) { hf[e] = 0.f; hl[e] = f2bf(0.f); }
  __syncthreads();

  const size_t xstride = (size_t)SS * G3;         // per-batch-row stride in Xp
  size_t xrow = (size_t)(bb + grow) * xstride + gd0; // step 0 address base
  size_t orow = ((size_t)(bb + grow) * SS) * 128 + gd0;

  for (int s = 0; s < SS; ++s) {
    // (1) issue this step's Xp loads up front — consumed only after the barrier
    float xr[8], xz[8], xn[8];
#pragma unroll
    for (int i = 0; i < 8; ++i) {
      xr[i] = Xp[xrow + i];
      xz[i] = Xp[xrow + 128 + i];
      xn[i] = Xp[xrow + 256 + i];
    }
    // (2) prefetch next step's gate rows into cache
    if (s + 1 < SS) {
      __builtin_prefetch(Xp + xrow + G3, 0, 1);
      __builtin_prefetch(Xp + xrow + G3 + 128, 0, 1);
      __builtin_prefetch(Xp + xrow + G3 + 256, 0, 1);
    }
    // (3) gh = h @ W_hh^T : 16x384x128 per step, 12 WMMAs per wave
    v16bf A[4];
    const bf16_t* arow = hl + m * 128 + half * 16;
#pragma unroll
    for (int kk = 0; kk < 4; ++kk) A[kk] = load_a_bf(arow + kk * 32);
#pragma unroll
    for (int j = 0; j < 3; ++j) {
      v8f acc = {0.f, 0.f, 0.f, 0.f, 0.f, 0.f, 0.f, 0.f};
#pragma unroll
      for (int kk = 0; kk < 4; ++kk)
        acc = __builtin_amdgcn_wmma_f32_16x16x32_bf16(false, A[kk], false, Breg[j][kk], (short)0, acc, false, false);
      int n = (wave * 3 + j) * 16 + m;
      int rb = half * 8;
#pragma unroll
      for (int r = 0; r < 8; ++r) ghl[(rb + r) * G3 + n] = acc[r];
    }
    __syncthreads();
    // (4) gate math on prefetched registers
#pragma unroll
    for (int i = 0; i < 8; ++i) {
      int d = gd0 + i;
      float hr = ghl[grow * G3 + d] + bhr[i];
      float hz = ghl[grow * G3 + 128 + d] + bhz[i];
      float hn = ghl[grow * G3 + 256 + d] + bhn[i];
      float r = sigm(xr[i] + hr);
      float z = sigm(xz[i] + hz);
      float nn = tanhf(xn[i] + r * hn);
      float hnew = (1.f - z) * nn + z * hf[grow * 128 + d];
      hf[grow * 128 + d] = hnew;
      hl[grow * 128 + d] = f2bf(hnew);
      rnnf[orow + i] = hnew;
      rnnb[orow + i] = f2bf(hnew);
    }
    __syncthreads();
    xrow += G3;
    orow += 128;
  }
}

// ---- kernel 5: s = (tanh(rnn @ mlp_w^T + mlp_b)) @ sim_w  via WMMA + fused reduce ----
__global__ void k_score(const bf16_t* __restrict__ rnnb, const bf16_t* __restrict__ mlpwb,
                        const float* __restrict__ mlpb, const float* __restrict__ simw,
                        float* __restrict__ scores) {
  int mt = (blockIdx.x * blockDim.x + threadIdx.x) >> 5;
  int lane = threadIdx.x & 31;
  int m = lane & 15, half = lane >> 4;

  v16bf A[4];
  const bf16_t* arow = rnnb + (size_t)(mt * 16 + m) * 128 + half * 16;
#pragma unroll
  for (int kk = 0; kk < 4; ++kk) A[kk] = load_a_bf(arow + kk * 32);

  float sacc[8] = {0.f, 0.f, 0.f, 0.f, 0.f, 0.f, 0.f, 0.f};
#pragma unroll
  for (int nt = 0; nt < 8; ++nt) {
    int n = nt * 16 + m;
    const bf16_t* bcol = mlpwb + (size_t)n * 128 + half * 16;
    v8f acc = {0.f, 0.f, 0.f, 0.f, 0.f, 0.f, 0.f, 0.f};
#pragma unroll
    for (int kk = 0; kk < 4; ++kk) {
      v16bf Bm = load_b_bf(bcol + kk * 32);
      acc = __builtin_amdgcn_wmma_f32_16x16x32_bf16(false, A[kk], false, Bm, (short)0, acc, false, false);
    }
    float bn = mlpb[n], wn = simw[n];
#pragma unroll
    for (int r = 0; r < 8; ++r) sacc[r] += tanhf(acc[r] + bn) * wn;
  }
  // reduce over the 16 lanes holding the 16 N-columns of each row
#pragma unroll
  for (int r = 0; r < 8; ++r) {
    float v = sacc[r];
    v += __shfl_xor(v, 1, 32);
    v += __shfl_xor(v, 2, 32);
    v += __shfl_xor(v, 4, 32);
    v += __shfl_xor(v, 8, 32);
    if (m == 0) scores[mt * 16 + half * 8 + r] = v;
  }
}

// ---- kernel 6: causal key-only softmax as an online running cumsum + exclusive prefix ----
__global__ void k_attn(const float* __restrict__ rnnf, const float* __restrict__ scores,
                       float* __restrict__ excl) {
  int b = blockIdx.x, d = threadIdx.x;
  float mx = -3.0e38f, den = 0.f, num = 0.f, eacc = 0.f;
  const float* sc = scores + (size_t)b * SS;
  const float* rp = rnnf + (size_t)b * SS * 128 + d;
  float* ep = excl + (size_t)b * SS * 128 + d;
  for (int i = 0; i < SS; ++i) {
    float s = sc[i];
    float nm = fmaxf(mx, s);
    float scale = __expf(mx - nm);
    float p = __expf(s - nm);
    num = num * scale + p * rp[(size_t)i * 128];
    den = den * scale + p;
    mx = nm;
    ep[(size_t)i * 128] = eacc;      // exclusive cumsum of attn
    eacc += num / den;
  }
}

// ---- kernel 7: prediction head: sigmoid(pred_w . [excl, rnn, concept(t+1)] + pred_b) ----
__global__ void k_pred(const float* __restrict__ excl, const float* __restrict__ rnnf,
                       const float* __restrict__ concept, const float* __restrict__ pw,
                       const float* __restrict__ pb, float* __restrict__ out) {
  int wave = (blockIdx.x * blockDim.x + threadIdx.x) >> 5;
  int lane = threadIdx.x & 31;
  int b = wave / (SS - 1), t = wave - b * (SS - 1);
  size_t row = (size_t)b * SS + t;
  float acc = 0.f;
#pragma unroll
  for (int it = 0; it < 10; ++it) {   // 320 = 10 * 32 lanes, uniform trip count
    int k = lane + it * 32;
    float x;
    if (k < 128)      x = excl[row * 128 + k];
    else if (k < 256) x = rnnf[row * 128 + (k - 128)];
    else              x = concept[(row + 1) * 64 + (k - 256)];
    acc += pw[k] * x;
  }
  acc += __shfl_xor(acc, 16, 32);
  acc += __shfl_xor(acc, 8, 32);
  acc += __shfl_xor(acc, 4, 32);
  acc += __shfl_xor(acc, 2, 32);
  acc += __shfl_xor(acc, 1, 32);
  if (lane == 0) out[(size_t)b * (SS - 1) + t] = sigm(acc + pb[0]);
}

extern "C" void kernel_launch(void* const* d_in, const int* in_sizes, int n_in,
                              void* d_out, int out_size, void* d_ws, size_t ws_size,
                              hipStream_t stream) {
  (void)in_sizes; (void)n_in; (void)out_size; (void)ws_size;
  const int*   qseq  = (const int*)d_in[0];
  const int*   cseq  = (const int*)d_in[1];
  const int*   q2c   = (const int*)d_in[2];
  const int*   q2cm  = (const int*)d_in[3];
  const float* cemb  = (const float*)d_in[4];
  const float* remb  = (const float*)d_in[5];
  const float* mlpw  = (const float*)d_in[6];
  const float* mlpb  = (const float*)d_in[7];
  const float* simw  = (const float*)d_in[8];
  const float* Wih   = (const float*)d_in[9];
  const float* bih   = (const float*)d_in[10];
  const float* Whh   = (const float*)d_in[11];
  const float* bhh   = (const float*)d_in[12];
  const float* pw    = (const float*)d_in[13];
  const float* pb    = (const float*)d_in[14];
  float* out = (float*)d_out;

  char* base = (char*)d_ws;
  auto alloc = [&](size_t bytes) -> void* {
    void* p = (void*)base;
    base += (bytes + 255) & ~(size_t)255;
    return p;
  };
  float*  concept = (float*) alloc((size_t)BS * 64 * 4);
  bf16_t* interb  = (bf16_t*)alloc((size_t)BS * 128 * 2);
  float*  Xp      = (float*) alloc((size_t)BS * G3 * 4);
  float*  rnnf    = (float*) alloc((size_t)BS * 128 * 4);
  bf16_t* rnnb    = (bf16_t*)alloc((size_t)BS * 128 * 2);
  float*  scores  = (float*) alloc((size_t)BS * 4);
  float*  excl    = (float*) alloc((size_t)BS * 128 * 4);
  bf16_t* Wihb    = (bf16_t*)alloc((size_t)G3 * 128 * 2);
  bf16_t* Whhb    = (bf16_t*)alloc((size_t)G3 * 128 * 2);
  bf16_t* mlpwb   = (bf16_t*)alloc((size_t)128 * 128 * 2);

  k_cvt  <<<192,  256, 0, stream>>>(Wih, Whh, mlpw, Wihb, Whhb, mlpwb);
  k_embed<<<8192, 256, 0, stream>>>(qseq, cseq, q2c, q2cm, cemb, remb, concept, interb);
  k_xp   <<<512,  256, 0, stream>>>(interb, Wihb, bih, Xp);        // 4096 M-tile waves
  k_gru  <<<4,    256, 0, stream>>>(Xp, Whhb, bhh, rnnf, rnnb);    // 4 x 16 batch rows
  k_score<<<512,  256, 0, stream>>>(rnnb, mlpwb, mlpb, simw, scores);
  k_attn <<<64,   128, 0, stream>>>(rnnf, scores, excl);
  k_pred <<<8184, 256, 0, stream>>>(excl, rnnf, concept, pw, pb, out);
}
